// CrossAttention_61959198212304
// MI455X (gfx1250) — matrix operand order repaired and checked by reading, not compiled, not run on hardware
//
#include <hip/hip_runtime.h>
#include <stdint.h>
#include <stddef.h>

// ---------------------------------------------------------------------------
// Problem constants (match reference)
// ---------------------------------------------------------------------------
#define BB   4
#define LQn  2048
#define LKn  2048
#define Dn   512
#define Hn   8
#define DHn  64
#define TOPKn 205

typedef __bf16 bf16;
typedef __attribute__((ext_vector_type(16))) __bf16 v16bf;
typedef __attribute__((ext_vector_type(8)))  float  v8f;

// ---------------------------------------------------------------------------
// gfx1250 async global->LDS staging (ASYNCcnt-tracked), guarded so a missing
// builtin falls back to register-staged loads + ds_store.
// Builtin signature (from probe diagnostic): param0 = AS(1) int4*, so:
//   (global int4*, lds int4*, imm offset, imm cpol)
// ---------------------------------------------------------------------------
#if defined(__has_builtin)
#if __has_builtin(__builtin_amdgcn_global_load_async_to_lds_b128) && \
    __has_builtin(__builtin_amdgcn_s_wait_asynccnt)
#define USE_ASYNC_LDS 1
#endif
#endif
#ifndef USE_ASYNC_LDS
#define USE_ASYNC_LDS 0
#endif

#if USE_ASYNC_LDS
typedef __attribute__((ext_vector_type(4))) int v4i;
typedef __attribute__((address_space(1))) v4i* gptr_v4i;
typedef __attribute__((address_space(3))) v4i* lptr_v4i;
__device__ __forceinline__ void async_b128(const bf16* g, bf16* l) {
  __builtin_amdgcn_global_load_async_to_lds_b128(
      (gptr_v4i)(g), (lptr_v4i)(l), 0, 0);
}
__device__ __forceinline__ void wait_async() {
  __builtin_amdgcn_s_wait_asynccnt(0);
}
#endif

__device__ __forceinline__ float wred_max(float v) {
  #pragma unroll
  for (int o = 16; o > 0; o >>= 1) v = fmaxf(v, __shfl_xor(v, o, 32));
  return v;
}
__device__ __forceinline__ float wred_sum(float v) {
  #pragma unroll
  for (int o = 16; o > 0; o >>= 1) v += __shfl_xor(v, o, 32);
  return v;
}
__device__ __forceinline__ int wred_sumi(int v) {
  #pragma unroll
  for (int o = 16; o > 0; o >>= 1) v += __shfl_xor(v, o, 32);
  return v;
}

// ---------------------------------------------------------------------------
// fp32 -> bf16 conversion
// ---------------------------------------------------------------------------
__global__ void cvt_f32_bf16(const float* __restrict__ in, bf16* __restrict__ out, size_t n) {
  size_t i = (size_t)blockIdx.x * blockDim.x + threadIdx.x;
  size_t stride = (size_t)gridDim.x * blockDim.x;
  for (; i < n; i += stride) out[i] = (bf16)in[i];
}

// ---------------------------------------------------------------------------
// Generic bf16 WMMA GEMM: C[M,N] = A[M,K] * W[N,K]^T + bias[N]
// 128x64 tile per WG (8 waves, each 32x32 = 2x2 WMMA frags), K-step 32,
// double-buffered LDS, fixed-trip staging (2x b128 for A, 1x b128 for W per
// thread per K-step).
// ---------------------------------------------------------------------------
#define GBM 128
#define GBN 64
#define GBK 32
#define LDT 48   // padded LDS stride (elems); 96B rows -> 32B-aligned subtiles

__global__ __launch_bounds__(256) void gemm_bf16_kernel(
    const bf16* __restrict__ A, const bf16* __restrict__ W,
    const float* __restrict__ bias, bf16* __restrict__ outb,
    float* __restrict__ outf, int M, int N, int K, int do_tanh)
{
  __shared__ __align__(32) bf16 As[2][GBM][LDT];
  __shared__ __align__(32) bf16 Ws[2][GBN][LDT];
  const int tid  = threadIdx.x;
  const int wave = tid >> 5, lane = tid & 31;
  const int half = lane >> 4, lr = lane & 15;
  const int bm = blockIdx.x * GBM;
  const int bn = blockIdx.y * GBN;
  const int wm = (wave & 3) * 32;   // 4 waves along M
  const int wn = (wave >> 2) * 32;  // 2 waves along N

  // staging map: 16B (8-elem) chunks; each thread owns fixed rows/cols
  const int srow = tid >> 2;          // 0..63
  const int scol = (tid & 3) * 8;     // 0,8,16,24
  const bf16* gA0 = A + (size_t)(bm + srow) * K + scol;
  const bf16* gA1 = A + (size_t)(bm + srow + 64) * K + scol;
  const bf16* gW  = W + (size_t)(bn + srow) * K + scol;
  const int nK = K / GBK;

  v8f c[2][2] = {};

  auto compute = [&](int buf) {
    v16bf a[2], bw[2];
    #pragma unroll
    for (int f = 0; f < 2; ++f) {
      const int m0 = wm + f * 16;
      #pragma unroll
      for (int i = 0; i < 16; ++i) {
        int kk = i + 16 * half;   // A 16x32 bf16 layout: contiguous k = 16*half + i
        a[f][i] = As[buf][m0 + lr][kk];
      }
      // B 32x16 layout: lane (half,lr) holds contiguous k = 16*half + i
      bw[f] = *(const v16bf*)&Ws[buf][wn + f * 16 + lr][16 * half];
    }
    #pragma unroll
    for (int fm = 0; fm < 2; ++fm)
      #pragma unroll
      for (int fn = 0; fn < 2; ++fn)
        c[fm][fn] = __builtin_amdgcn_wmma_f32_16x16x32_bf16(
            false, a[fm], false, bw[fn], (short)0, c[fm][fn], false, false);
  };

#if USE_ASYNC_LDS
  auto stage = [&](int kt, int buf) {
    const int k0 = kt * GBK;
    async_b128(gA0 + k0, &As[buf][srow][scol]);
    async_b128(gA1 + k0, &As[buf][srow + 64][scol]);
    async_b128(gW  + k0, &Ws[buf][srow][scol]);
  };
  stage(0, 0);
  wait_async();
  __syncthreads();
  for (int kt = 0; kt < nK; ++kt) {
    const int cur = kt & 1;
    if (kt + 1 < nK) stage(kt + 1, cur ^ 1);   // overlap with compute
    compute(cur);
    wait_async();
    __syncthreads();
  }
#else
  uint4 ra0, ra1, rw;
  auto gload = [&](int kt) {
    const int k0 = kt * GBK;
    ra0 = *(const uint4*)(gA0 + k0);
    ra1 = *(const uint4*)(gA1 + k0);
    rw  = *(const uint4*)(gW + k0);
  };
  auto sstore = [&](int buf) {
    *(uint4*)&As[buf][srow][scol]      = ra0;
    *(uint4*)&As[buf][srow + 64][scol] = ra1;
    *(uint4*)&Ws[buf][srow][scol]      = rw;
  };
  gload(0);
  sstore(0);
  __syncthreads();
  for (int kt = 0; kt < nK; ++kt) {
    const int cur = kt & 1;
    if (kt + 1 < nK) gload(kt + 1);            // loads in flight over compute
    compute(cur);
    if (kt + 1 < nK) sstore(cur ^ 1);
    __syncthreads();
  }
#endif

  // epilogue: bias (+tanh), store bf16 or fp32
  #pragma unroll
  for (int fm = 0; fm < 2; ++fm)
    #pragma unroll
    for (int fn = 0; fn < 2; ++fn)
      #pragma unroll
      for (int r = 0; r < 8; ++r) {
        int m = bm + wm + fm * 16 + r + 8 * half;  // C/D: M = r + 8*(lane/16)
        int n = bn + wn + fn * 16 + lr;
        float v = c[fm][fn][r] + bias[n];
        if (do_tanh) v = tanhf(v);
        if (outf) outf[(size_t)m * N + n] = v;
        else      outb[(size_t)m * N + n] = (bf16)v;
      }
}

// ---------------------------------------------------------------------------
// l2 normalize rows of 64 (one wave per row; 8 rows per block)
// ---------------------------------------------------------------------------
__global__ __launch_bounds__(256) void l2norm_kernel(bf16* __restrict__ x, int rows) {
  const int wave = threadIdx.x >> 5, lane = threadIdx.x & 31;
  const int row = blockIdx.x * 8 + wave;
  if (row >= rows) return;
  bf16* p = x + (size_t)row * DHn;
  float v0 = (float)p[lane * 2], v1 = (float)p[lane * 2 + 1];
  float s = wred_sum(v0 * v0 + v1 * v1);
  float sc = 1.0f / fmaxf(sqrtf(s), 1e-12f);
  p[lane * 2]     = (bf16)(v0 * sc);
  p[lane * 2 + 1] = (bf16)(v1 * sc);
}

// ---------------------------------------------------------------------------
// Attention: per WG one (b,h) x 16-row q tile.
//  phase 1: logits 16x2048 via WMMA into LDS fp32 (scaled, masked)
//  phase 2: softmax + smooth + gate + binary-search top-k threshold
//  phase 3: attn(16x2048) @ V(2048x64) via WMMA (double-buffered V staging)
// ---------------------------------------------------------------------------
#define QB 16
__global__ __launch_bounds__(256) void attn_kernel(
    const bf16* __restrict__ Qp, const bf16* __restrict__ Kp,
    const bf16* __restrict__ Vp, const unsigned char* __restrict__ mask,
    bf16* __restrict__ ctx)
{
  __shared__ float S[QB][LKn];                    // 131072 B
  __shared__ __align__(32) bf16 qst[QB][72];      // q tile staging
  __shared__ __align__(32) bf16 vst[2][32][80];   // double-buffered V staging

  const int tid = threadIdx.x, wave = tid >> 5, lane = tid & 31;
  const int half = lane >> 4, lr = lane & 15;
  const int blk = blockIdx.x;
  const int qb = blk & (LQn / QB - 1);            // 128 q-tiles
  const int h  = (blk >> 7) & (Hn - 1);
  const int b  = blk >> 10;
  const int q0 = qb * QB;
  const float SCALE = 0.125f;                     // 1/sqrt(64), ATTN_TEMP=1
  const float EPS_K = 0.01f / (float)LKn;

  // ---- stage Q tile (16 x 64): 128 b128 chunks, first 128 threads ----
  if (tid < 128) {
    const int r = tid >> 3, col = (tid & 7) * 8;
    *(uint4*)&qst[r][col] =
        *(const uint4*)&Qp[((size_t)(b * LQn + q0 + r) * Hn + h) * DHn + col];
  }
  __syncthreads();

  // ---- phase 1: logits. wave w owns columns [w*256, w*256+256) ----
  v16bf aq[2];
  #pragma unroll
  for (int ef = 0; ef < 2; ++ef)
    #pragma unroll
    for (int i = 0; i < 16; ++i) {
      int kk = ef * 32 + i + 16 * half;
      aq[ef][i] = qst[lr][kk];
    }

  for (int nt = 0; nt < 16; ++nt) {
    const int n0 = wave * 256 + nt * 16;
    if (nt + 1 < 16) {  // prefetch next tile's K row for this lane
      __builtin_prefetch(
          &Kp[((size_t)(b * LKn + n0 + 16 + lr) * Hn + h) * DHn + 16 * half], 0, 3);
    }
    v8f c = {};
    #pragma unroll
    for (int ef = 0; ef < 2; ++ef) {
      // B-frag: k-dim = head dim e; lane's 16 elems are contiguous 32B in K row
      v16bf bk = *(const v16bf*)&Kp[((size_t)(b * LKn + n0 + lr) * Hn + h) * DHn
                                    + ef * 32 + 16 * half];
      c = __builtin_amdgcn_wmma_f32_16x16x32_bf16(
          false, aq[ef], false, bk, (short)0, c, false, false);
    }
    #pragma unroll
    for (int r = 0; r < 8; ++r) {
      int m = r + 8 * half;
      int n = n0 + lr;
      float v = c[r] * SCALE;
      if (mask[(size_t)b * LKn + n]) v = -10000.0f;
      S[m][n] = v;
    }
  }
  __syncthreads();

  // ---- phase 2: softmax + smooth + gate + top-k threshold (2 rows / wave) ----
  for (int rr = 0; rr < 2; ++rr) {
    const int row = wave * 2 + rr;
    float av[64];
    float mx = -1e30f;
    #pragma unroll
    for (int j = 0; j < 64; ++j) { av[j] = S[row][lane + 32 * j]; mx = fmaxf(mx, av[j]); }
    mx = wred_max(mx);
    float sum = 0.f;
    #pragma unroll
    for (int j = 0; j < 64; ++j) { av[j] = __expf(av[j] - mx); sum += av[j]; }
    sum = wred_sum(sum);
    const float inv = 1.0f / sum;
    float amax = 0.f;
    #pragma unroll
    for (int j = 0; j < 64; ++j) {
      float p = av[j] * inv;
      float a = 0.99f * p + EPS_K;          // label smoothing
      a = (a >= 1e-4f) ? a : 0.0f;          // hard gate
      av[j] = a;
      amax = fmaxf(amax, a);
    }
    amax = wred_max(amax);
    float lo = 0.f, hi = amax;              // binary search TOPK-th value
    for (int it = 0; it < 24; ++it) {
      float mid = 0.5f * (lo + hi);
      int cnt = 0;
      #pragma unroll
      for (int j = 0; j < 64; ++j) cnt += (av[j] >= mid) ? 1 : 0;
      cnt = wred_sumi(cnt);
      if (cnt > TOPKn) lo = mid; else hi = mid;
    }
    const float thr = hi;
    #pragma unroll
    for (int j = 0; j < 64; ++j)
      S[row][lane + 32 * j] = (av[j] >= thr) ? av[j] : 0.0f;
  }
  __syncthreads();

  // ---- phase 3: ctx = attn @ V, waves 0..3 own one 16-wide n-tile each ----
  const int vrow = tid >> 3, vcol = (tid & 7) * 8;     // 16B chunk per thread
  const bf16* gV = Vp + ((size_t)(b * LKn + vrow) * Hn + h) * DHn + vcol;
  const size_t vstep = (size_t)32 * Hn * DHn;          // 32 k-rows per step
  v8f cacc = {};
  const int myn = wave & 3;
  const bool active = (wave < 4);
  const int nKB = LKn / 32;

  auto vcompute = [&](int buf, int k0) {
    v16bf af, bv;
    #pragma unroll
    for (int i = 0; i < 16; ++i) {
      int kk = i + 16 * half;
      af[i] = (bf16)S[lr][k0 + kk];
      bv[i] = vst[buf][i + 16 * half][myn * 16 + lr];
    }
    cacc = __builtin_amdgcn_wmma_f32_16x16x32_bf16(
        false, af, false, bv, (short)0, cacc, false, false);
  };

#if USE_ASYNC_LDS
  async_b128(gV, &vst[0][vrow][vcol]);
  wait_async();
  __syncthreads();
  for (int kb = 0; kb < nKB; ++kb) {
    const int cur = kb & 1;
    if (kb + 1 < nKB) async_b128(gV + (size_t)(kb + 1) * vstep, &vst[cur ^ 1][vrow][vcol]);
    if (active) vcompute(cur, kb * 32);
    wait_async();
    __syncthreads();
  }
#else
  uint4 rv = *(const uint4*)gV;
  *(uint4*)&vst[0][vrow][vcol] = rv;
  __syncthreads();
  for (int kb = 0; kb < nKB; ++kb) {
    const int cur = kb & 1;
    if (kb + 1 < nKB) rv = *(const uint4*)(gV + (size_t)(kb + 1) * vstep);
    if (active) vcompute(cur, kb * 32);
    if (kb + 1 < nKB) *(uint4*)&vst[cur ^ 1][vrow][vcol] = rv;
    __syncthreads();
  }
#endif

  if (active) {
    #pragma unroll
    for (int r = 0; r < 8; ++r) {
      int m = r + 8 * half;
      int n = myn * 16 + lr;
      // merged-head layout [B, LQ, H*DH]
      ctx[(size_t)(b * LQn + q0 + m) * Dn + h * DHn + n] = (bf16)cacc[r];
    }
  }
}

// ---------------------------------------------------------------------------
// Host orchestration
// ---------------------------------------------------------------------------
extern "C" void kernel_launch(void* const* d_in, const int* in_sizes, int n_in,
                              void* d_out, int out_size, void* d_ws, size_t ws_size,
                              hipStream_t stream) {
  (void)in_sizes; (void)n_in; (void)out_size; (void)ws_size;
  const float* q_in = (const float*)d_in[0];
  const float* k_in = (const float*)d_in[1];
  const float* v_in = (const float*)d_in[2];
  const float* Wq   = (const float*)d_in[3];
  const float* bq   = (const float*)d_in[4];
  const float* Wk   = (const float*)d_in[5];
  const float* bk   = (const float*)d_in[6];
  const float* Wv   = (const float*)d_in[7];
  const float* bv   = (const float*)d_in[8];
  const float* Wo   = (const float*)d_in[9];
  const float* bo   = (const float*)d_in[10];
  const unsigned char* mask = (const unsigned char*)d_in[11];
  float* out = (float*)d_out;

  const size_t M   = (size_t)BB * LQn;     // 8192 tokens
  const size_t ACT = M * Dn;               // 4,194,304 elements
  const size_t WEL = (size_t)Dn * Dn;      // 262,144 elements

  char* ws = (char*)d_ws;
  size_t off = 0;
  auto alloc = [&](size_t bytes) -> void* {
    void* p = ws + off;
    off = (off + bytes + 255) & ~(size_t)255;
    return p;
  };
  bf16* qin_b = (bf16*)alloc(ACT * 2);
  bf16* kin_b = (bf16*)alloc(ACT * 2);
  bf16* vin_b = (bf16*)alloc(ACT * 2);
  bf16* wq_b  = (bf16*)alloc(WEL * 2);
  bf16* wk_b  = (bf16*)alloc(WEL * 2);
  bf16* wv_b  = (bf16*)alloc(WEL * 2);
  bf16* wo_b  = (bf16*)alloc(WEL * 2);
  bf16* qproj = (bf16*)alloc(ACT * 2);
  bf16* kproj = (bf16*)alloc(ACT * 2);
  bf16* vproj = (bf16*)alloc(ACT * 2);
  bf16* ctxb  = (bf16*)alloc(ACT * 2);

  // convert activations + weights to bf16
  cvt_f32_bf16<<<2048, 256, 0, stream>>>(q_in, qin_b, ACT);
  cvt_f32_bf16<<<2048, 256, 0, stream>>>(k_in, kin_b, ACT);
  cvt_f32_bf16<<<2048, 256, 0, stream>>>(v_in, vin_b, ACT);
  cvt_f32_bf16<<<512, 256, 0, stream>>>(Wq, wq_b, WEL);
  cvt_f32_bf16<<<512, 256, 0, stream>>>(Wk, wk_b, WEL);
  cvt_f32_bf16<<<512, 256, 0, stream>>>(Wv, wv_b, WEL);
  cvt_f32_bf16<<<512, 256, 0, stream>>>(Wo, wo_b, WEL);

  // per-head projections as one big GEMM each (N dim = h*64+e, h-major)
  dim3 gg(M / GBM, Dn / GBN);
  gemm_bf16_kernel<<<gg, 256, 0, stream>>>(qin_b, wq_b, bq, qproj, nullptr,
                                           (int)M, Dn, Dn, 0);
  gemm_bf16_kernel<<<gg, 256, 0, stream>>>(kin_b, wk_b, bk, kproj, nullptr,
                                           (int)M, Dn, Dn, 0);
  gemm_bf16_kernel<<<gg, 256, 0, stream>>>(vin_b, wv_b, bv, vproj, nullptr,
                                           (int)M, Dn, Dn, 1 /*tanh*/);

  // qk-norm over DH=64 head rows
  const int headRows = (int)(M * Hn);      // 65536
  l2norm_kernel<<<headRows / 8, 256, 0, stream>>>(qproj, headRows);
  l2norm_kernel<<<headRows / 8, 256, 0, stream>>>(kproj, headRows);

  // attention (softmax + smooth + gate + top-k + attn@V)
  attn_kernel<<<BB * Hn * (LQn / QB), 256, 0, stream>>>(qproj, kproj, vproj,
                                                        mask, ctxb);

  // output projection -> fp32 d_out
  gemm_bf16_kernel<<<gg, 256, 0, stream>>>(ctxb, wo_b, bo, nullptr, out,
                                           (int)M, Dn, Dn, 0);
}
